// Encoder_78099685310768
// MI455X (gfx1250) — matrix operand order repaired and checked by reading, not compiled, not run on hardware
//
#include <hip/hip_runtime.h>
#include <stdint.h>

typedef __attribute__((ext_vector_type(16))) __bf16   v16bf;
typedef __attribute__((ext_vector_type(8)))  float    v8f;
typedef __attribute__((ext_vector_type(4)))  uint32_t u32x4;

// ---------- helpers ----------
static __device__ __forceinline__ uint16_t f2bf(float x) {
  union { float f; uint32_t u; } c; c.f = x;
  uint32_t r = c.u + 0x7FFFu + ((c.u >> 16) & 1u);
  return (uint16_t)(r >> 16);
}
static __device__ __forceinline__ float bf2f(uint16_t h) {
  union { float f; uint32_t u; } c; c.u = ((uint32_t)h) << 16; return c.f;
}

union FragCast { u32x4 q[2]; v16bf v; };

// A fragment (16 rows x 32 K), row-major bf16 source, lda multiple of 32.
// Lanes 0-15: row = row0+lane, K {k0..k0+7, k0+16..k0+23}
// Lanes 16-31: row = row0+lane-16, K {k0+8..k0+15, k0+24..k0+31}
static __device__ __forceinline__ v16bf load_frag_a(const uint16_t* __restrict__ A,
                                                    int lda, int row0, int k0, int lane) {
  const uint16_t* p = A + (size_t)(row0 + (lane & 15)) * lda + (size_t)(k0 + ((lane >> 4) << 3));
  FragCast u;
  u.q[0] = *reinterpret_cast<const u32x4*>(p);
  u.q[1] = *reinterpret_cast<const u32x4*>(p + 16);
  return u.v;
}
// B fragment (32 K x 16 cols) from B^T (cols stored as rows), ldb multiple of 32.
// Lanes 0-15: col = col0+lane, K {k0..k0+7, k0+16..k0+23} ; lanes 16-31: K {k0+8..k0+15, k0+24..k0+31}
static __device__ __forceinline__ v16bf load_frag_b(const uint16_t* __restrict__ BT,
                                                    int ldb, int col0, int k0, int lane) {
  const uint16_t* p = BT + (size_t)(col0 + (lane & 15)) * ldb + (size_t)(k0 + ((lane >> 4) << 3));
  FragCast u;
  u.q[0] = *reinterpret_cast<const u32x4*>(p);
  u.q[1] = *reinterpret_cast<const u32x4*>(p + 16);
  return u.v;
}

// ---------- generic bf16 WMMA GEMM ----------
// C(M,N) = A(M,K) * B(K,N);  A row-major bf16, B given as BT[N][K] bf16.
// Epilogue: + colbias[col%cmod] + bias2d[row*cmod + col%cmod]; optional PReLU(alpha[col%cmod]).
// mode 0: outF[row*ldo+col] f32 (row-major). mode 1: outT[col*ldo+row] bf16 (transposed).
__global__ __launch_bounds__(256) void k_gemm_bf16(
    const uint16_t* __restrict__ A, int lda,
    const uint16_t* __restrict__ BT, int ldb, int K,
    float* __restrict__ outF, uint16_t* __restrict__ outT, int ldo,
    const float* __restrict__ colbias, int cmod,
    const float* __restrict__ alpha,
    const float* __restrict__ bias2d, int mode)
{
  const int lane = threadIdx.x & 31;
  const int wave = threadIdx.x >> 5;
  const int wm = (wave & 1) * 32;          // 2 wave rows  -> block M tile 64
  const int wn = (wave >> 1) * 64;         // 4 wave cols  -> block N tile 256
  const int bm = blockIdx.y * 64;
  const int bn = blockIdx.x * 256;

  const v8f vzero = {0.f,0.f,0.f,0.f,0.f,0.f,0.f,0.f};
  v8f acc[2][4];
  #pragma unroll
  for (int i = 0; i < 2; ++i)
    #pragma unroll
    for (int j = 0; j < 4; ++j) acc[i][j] = vzero;

  for (int k = 0; k < K; k += 32) {
    v16bf a0 = load_frag_a(A, lda, bm + wm,      k, lane);
    v16bf a1 = load_frag_a(A, lda, bm + wm + 16, k, lane);
    v16bf b0 = load_frag_b(BT, ldb, bn + wn,      k, lane);
    v16bf b1 = load_frag_b(BT, ldb, bn + wn + 16, k, lane);
    v16bf b2 = load_frag_b(BT, ldb, bn + wn + 32, k, lane);
    v16bf b3 = load_frag_b(BT, ldb, bn + wn + 48, k, lane);
    acc[0][0] = __builtin_amdgcn_wmma_f32_16x16x32_bf16(false, a0, false, b0, (short)0, acc[0][0], false, false);
    acc[0][1] = __builtin_amdgcn_wmma_f32_16x16x32_bf16(false, a0, false, b1, (short)0, acc[0][1], false, false);
    acc[0][2] = __builtin_amdgcn_wmma_f32_16x16x32_bf16(false, a0, false, b2, (short)0, acc[0][2], false, false);
    acc[0][3] = __builtin_amdgcn_wmma_f32_16x16x32_bf16(false, a0, false, b3, (short)0, acc[0][3], false, false);
    acc[1][0] = __builtin_amdgcn_wmma_f32_16x16x32_bf16(false, a1, false, b0, (short)0, acc[1][0], false, false);
    acc[1][1] = __builtin_amdgcn_wmma_f32_16x16x32_bf16(false, a1, false, b1, (short)0, acc[1][1], false, false);
    acc[1][2] = __builtin_amdgcn_wmma_f32_16x16x32_bf16(false, a1, false, b2, (short)0, acc[1][2], false, false);
    acc[1][3] = __builtin_amdgcn_wmma_f32_16x16x32_bf16(false, a1, false, b3, (short)0, acc[1][3], false, false);
  }

  // Epilogue. C/D layout: VGPR r, lanes 0-15 -> (row tile+r, col tile+lane);
  // lanes 16-31 -> (row tile+8+r, col tile+lane-16).
  #pragma unroll
  for (int tm = 0; tm < 2; ++tm) {
    #pragma unroll
    for (int tn = 0; tn < 4; ++tn) {
      const int col   = bn + wn + tn * 16 + (lane & 15);
      const int rbase = bm + wm + tm * 16 + ((lane >> 4) << 3);
      const int f = col % cmod;
      const float cb = colbias ? colbias[f] : 0.0f;
      float vals[8];
      #pragma unroll
      for (int r = 0; r < 8; ++r) {
        float v = acc[tm][tn][r] + cb;
        if (bias2d) v += bias2d[(size_t)(rbase + r) * cmod + f];
        if (alpha)  { float al = alpha[f]; v = (v >= 0.0f) ? v : al * v; }
        vals[r] = v;
      }
      if (mode == 0) {
        #pragma unroll
        for (int r = 0; r < 8; ++r)
          outF[(size_t)(rbase + r) * ldo + col] = vals[r];
      } else {
        u32x4 st;
        st[0] = (uint32_t)f2bf(vals[0]) | ((uint32_t)f2bf(vals[1]) << 16);
        st[1] = (uint32_t)f2bf(vals[2]) | ((uint32_t)f2bf(vals[3]) << 16);
        st[2] = (uint32_t)f2bf(vals[4]) | ((uint32_t)f2bf(vals[5]) << 16);
        st[3] = (uint32_t)f2bf(vals[6]) | ((uint32_t)f2bf(vals[7]) << 16);
        *reinterpret_cast<u32x4*>(outT + (size_t)col * ldo + rbase) = st;
      }
    }
  }
}

// ---------- conversions / small kernels ----------
__global__ void k_f32_to_bf16(const float* __restrict__ s, uint16_t* __restrict__ d, long n) {
  long i = (long)blockIdx.x * blockDim.x + threadIdx.x;
  if (i < n) d[i] = f2bf(s[i]);
}
__global__ void k_zero_bf16(uint16_t* __restrict__ d, long n) {
  long i = (long)blockIdx.x * blockDim.x + threadIdx.x;
  if (i < n) d[i] = 0;
}
// WxT[u][d] = Wx[d][u] (bf16), K padded 4097 -> 4128 with zeros
__global__ void k_build_wxt(const float* __restrict__ Wx, uint16_t* __restrict__ WxT) {
  long g = (long)blockIdx.x * blockDim.x + threadIdx.x;
  if (g >= (long)1024 * 4128) return;
  int u = (int)(g / 4128), d = (int)(g % 4128);
  float v = (d < 4097) ? Wx[(size_t)d * 1024 + u] : 0.0f;
  WxT[(size_t)u * 4128 + d] = f2bf(v);
}
// channel mix: tT[(b*Cout+fo)][m] = sum_c in[b,m,c] * W[c][fo]
// in = feat0 (f32, [b][m][c]) for level 0, else vT (bf16, [(b*Cin+c)][m])
__global__ void k_stepA(const uint16_t* __restrict__ vT, const float* __restrict__ feat0,
                        const float* __restrict__ W, int Cin, int Cout, int Nrow,
                        uint16_t* __restrict__ tT)
{
  int m = blockIdx.x * blockDim.x + threadIdx.x;
  int j = blockIdx.y;
  if (m >= Nrow) return;
  int b = j / Cout, fo = j % Cout;
  float acc = 0.0f;
  for (int c = 0; c < Cin; ++c) {
    float x = feat0 ? feat0[((size_t)b * Nrow + m) * Cin + c]
                    : bf2f(vT[(size_t)(b * Cin + c) * Nrow + m]);
    acc = fmaf(x, W[c * Cout + fo], acc);
  }
  tT[(size_t)j * Nrow + m] = f2bf(acc);
}
// Ragged scatter: x_pad[sid*30+pos][p*64+f] = v2[p][b*64+f]; [4096]=times[b]
__global__ void k_pack_x(const float* __restrict__ v2, const float* __restrict__ times,
                         const int* __restrict__ rl, uint16_t* __restrict__ xp)
{
  int g = blockIdx.x * blockDim.x + threadIdx.x;
  if (g >= 800 * 64) return;
  int b = g >> 6, p = g & 63;
  int cum = 0, sid = 0, pos = 0;
  for (int s = 0; s < 32; ++s) {
    int nx = cum + rl[s];
    if (b < nx) { sid = s; pos = b - cum; break; }
    cum = nx;
  }
  int row = sid * 30 + pos;
  uint16_t* xr = xp + (size_t)row * 4128 + p * 64;
  const float* vr = v2 + (size_t)p * 51200 + (size_t)b * 64;
  for (int f = 0; f < 64; ++f) xr[f] = f2bf(vr[f]);
  if (p == 0) xp[(size_t)row * 4128 + 4096] = f2bf(times[b]);
}

// ---------- LSTM recurrence + FC + MLP head (one workgroup per sequence) ----------
__global__ __launch_bounds__(1024) void k_lstm_head(
    const float* __restrict__ Xz, const int* __restrict__ rl,
    const float* __restrict__ Wh,
    const float* __restrict__ Wfc, const float* __restrict__ bfc,
    const float* __restrict__ Ws1, const float* __restrict__ bs1,
    const float* __restrict__ Ws2, const float* __restrict__ bs2,
    float* __restrict__ out)
{
  __shared__ float sh[256], sc[256], sz[1024], sfc[130], ss1[256];
  const int tid = threadIdx.x;
  const int s = blockIdx.x;
  if (tid < 256) { sh[tid] = 0.f; sc[tid] = 0.f; }
  __syncthreads();
  const int len = rl[s];
  for (int t = 0; t < len; ++t) {
    // z[col] = Xz(row, col) + sum_k h[k]*Wh[k][col]   (col = gate*256 + u; Xz already has +bl)
    float acc = Xz[(size_t)(s * 30 + t) * 1024 + tid];
    const float* w = Wh + tid;
    for (int k = 0; k < 256; ++k)
      acc = fmaf(sh[k], w[(size_t)k * 1024], acc);
    sz[tid] = acc;
    __syncthreads();
    if (tid < 256) {
      float zi = sz[tid], zf = sz[256 + tid], zg = sz[512 + tid], zo = sz[768 + tid];
      float ig = 1.f / (1.f + expf(-zi));
      float fg = 1.f / (1.f + expf(-zf));
      float gg = tanhf(zg);
      float og = 1.f / (1.f + expf(-zo));
      float cn = fg * sc[tid] + ig * gg;
      sc[tid] = cn;
      sh[tid] = og * tanhf(cn);
    }
    __syncthreads();
  }
  // FC head
  if (tid < 130) {
    float acc = bfc[tid];
    for (int k = 0; k < 256; ++k) acc = fmaf(sh[k], Wfc[k * 130 + tid], acc);
    sfc[tid] = acc;
  }
  __syncthreads();
  if (tid < 256) {
    float acc = bs1[tid];
    for (int k = 0; k < 128; ++k) acc = fmaf(sfc[2 + k], Ws1[k * 256 + tid], acc);
    ss1[tid] = fmaxf(acc, 0.f);
  }
  __syncthreads();
  if (tid < 64) {
    float acc = bs2[tid];
    for (int k = 0; k < 256; ++k) acc = fmaf(ss1[k], Ws2[k * 64 + tid], acc);
    out[s * 66 + 2 + tid] = acc;
  }
  if (tid == 64) { out[s * 66 + 0] = sfc[0]; out[s * 66 + 1] = sfc[1]; }
}

// ---------- host ----------
extern "C" void kernel_launch(void* const* d_in, const int* in_sizes, int n_in,
                              void* d_out, int out_size, void* d_ws, size_t ws_size,
                              hipStream_t stream)
{
  (void)in_sizes; (void)n_in; (void)out_size; (void)ws_size;
  const float* features = (const float*)d_in[0];
  const float* times    = (const float*)d_in[1];
  const int*   rl       = (const int*)d_in[2];
  const float* L[3]  = { (const float*)d_in[3],  (const float*)d_in[9],  (const float*)d_in[15] };
  const float* W[3]  = { (const float*)d_in[4],  (const float*)d_in[10], (const float*)d_in[16] };
  const float* bb[3] = { (const float*)d_in[5],  (const float*)d_in[11], (const float*)d_in[17] };
  const float* aa[3] = { (const float*)d_in[6],  (const float*)d_in[12], (const float*)d_in[18] };
  const float* D[3]  = { (const float*)d_in[7],  (const float*)d_in[13], (const float*)d_in[19] };
  const float* Db[3] = { (const float*)d_in[8],  (const float*)d_in[14], (const float*)d_in[20] };
  const float* Wx  = (const float*)d_in[21];
  const float* Wh  = (const float*)d_in[22];
  const float* bl  = (const float*)d_in[23];
  const float* Wfc = (const float*)d_in[24];
  const float* bfc = (const float*)d_in[25];
  const float* Ws1 = (const float*)d_in[26];
  const float* bs1 = (const float*)d_in[27];
  const float* Ws2 = (const float*)d_in[28];
  const float* bs2 = (const float*)d_in[29];
  float* out = (float*)d_out;

  char* base = (char*)d_ws;
  size_t off = 0;
  auto alloc = [&](size_t bytes) -> char* {
    char* p = base + off;
    off += (bytes + 255) & ~(size_t)255;
    return p;
  };
  uint16_t* Lb[3]; uint16_t* DbF[3];
  Lb[0]  = (uint16_t*)alloc((size_t)4096*4096*2);
  Lb[1]  = (uint16_t*)alloc((size_t)1024*1024*2);
  Lb[2]  = (uint16_t*)alloc((size_t)256*256*2);
  DbF[0] = (uint16_t*)alloc((size_t)1024*4096*2);
  DbF[1] = (uint16_t*)alloc((size_t)256*1024*2);
  DbF[2] = (uint16_t*)alloc((size_t)64*256*2);
  uint16_t* WxT  = (uint16_t*)alloc((size_t)1024*4128*2);
  uint16_t* xpad = (uint16_t*)alloc((size_t)960*4128*2);
  float*    Xz   = (float*)   alloc((size_t)960*1024*4);
  char* poolA = alloc((size_t)12800*4096*2);   // 104.9 MB
  char* poolB = alloc((size_t)12800*4096*2);   // 104.9 MB

  auto conv = [&](const float* s, uint16_t* d, long n) {
    k_f32_to_bf16<<<dim3((unsigned)((n + 255) / 256)), dim3(256), 0, stream>>>(s, d, n);
  };
  conv(L[0], Lb[0], (long)4096*4096);
  conv(L[1], Lb[1], (long)1024*1024);
  conv(L[2], Lb[2], (long)256*256);
  conv(D[0], DbF[0], (long)1024*4096);
  conv(D[1], DbF[1], (long)256*1024);
  conv(D[2], DbF[2], (long)64*256);
  k_build_wxt<<<dim3((unsigned)(((long)1024*4128 + 255) / 256)), dim3(256), 0, stream>>>(Wx, WxT);

  auto gemm = [&](const uint16_t* A, int lda, const uint16_t* BT, int ldb,
                  int M, int N, int K, float* oF, uint16_t* oT, int ldo,
                  const float* cb, int cmod, const float* al, const float* b2d, int mode) {
    dim3 g((unsigned)(N / 256), (unsigned)(M / 64));
    k_gemm_bf16<<<g, dim3(256), 0, stream>>>(A, lda, BT, ldb, K, oF, oT, ldo, cb, cmod, al, b2d, mode);
  };
  auto stepA = [&](const uint16_t* vT, const float* feat0, const float* Wm,
                   int Cin, int Cout, int Nrow, int B, uint16_t* tT) {
    dim3 g((unsigned)((Nrow + 255) / 256), (unsigned)(B * Cout));
    k_stepA<<<g, dim3(256), 0, stream>>>(vT, feat0, Wm, Cin, Cout, Nrow, tT);
  };

  const int Bt = 800;
  // ---- level 0: N=4096 -> 1024, C 3->16 ----
  uint16_t* t0 = (uint16_t*)poolA;             // (12800 x 4096)
  uint16_t* u0 = (uint16_t*)poolB;             // (12800 x 4096)
  uint16_t* v0 = (uint16_t*)poolA;             // (12800 x 1024)
  stepA(nullptr, features, W[0], 3, 16, 4096, Bt, t0);
  gemm(Lb[0], 4096, t0, 4096, 4096, 12800, 4096, nullptr, u0, 4096, bb[0], 16, aa[0], nullptr, 1);
  gemm(DbF[0], 4096, u0, 4096, 1024, 12800, 4096, nullptr, v0, 1024, nullptr, 16, nullptr, Db[0], 1);
  // ---- level 1: N=1024 -> 256, C 16->32 ----
  uint16_t* t1 = (uint16_t*)poolB;             // (25600 x 1024)
  uint16_t* u1 = (uint16_t*)poolA;             // (25600 x 1024)
  uint16_t* v1 = (uint16_t*)poolB;             // (25600 x 256)
  stepA(v0, nullptr, W[1], 16, 32, 1024, Bt, t1);
  gemm(Lb[1], 1024, t1, 1024, 1024, 25600, 1024, nullptr, u1, 1024, bb[1], 32, aa[1], nullptr, 1);
  gemm(DbF[1], 1024, u1, 1024, 256, 25600, 1024, nullptr, v1, 256, nullptr, 32, nullptr, Db[1], 1);
  // ---- level 2: N=256 -> 64, C 32->64 ----
  uint16_t* t2 = (uint16_t*)poolA;             // (51200 x 256)
  uint16_t* u2 = (uint16_t*)poolB;             // (51200 x 256)
  float*    v2 = (float*)poolA;                // (64 x 51200) f32 row-major
  stepA(v1, nullptr, W[2], 32, 64, 256, Bt, t2);
  gemm(Lb[2], 256, t2, 256, 256, 51200, 256, nullptr, u2, 256, bb[2], 64, aa[2], nullptr, 1);
  gemm(DbF[2], 256, u2, 256, 64, 51200, 256, v2, nullptr, 51200, nullptr, 64, nullptr, Db[2], 0);
  // ---- ragged pack into padded LSTM input ----
  k_zero_bf16<<<dim3((unsigned)(((long)960*4128 + 255) / 256)), dim3(256), 0, stream>>>(xpad, (long)960*4128);
  k_pack_x<<<dim3((unsigned)((800*64 + 255) / 256)), dim3(256), 0, stream>>>(v2, times, rl, xpad);
  // ---- LSTM input GEMM: Xz = xpad @ Wx + bl  (K padded to 4128) ----
  gemm(xpad, 4128, WxT, 4128, 960, 1024, 4128, Xz, nullptr, 1024, bl, 1024, nullptr, nullptr, 0);
  // ---- recurrence + heads ----
  k_lstm_head<<<dim3(32), dim3(1024), 0, stream>>>(Xz, rl, Wh, Wfc, bfc, Ws1, bs1, Ws2, bs2, out);
}
